// EncoderLayer_29454885715997
// MI455X (gfx1250) — matrix operand order repaired and checked by reading, not compiled, not run on hardware
//
#include <hip/hip_runtime.h>
#include <hip/hip_bf16.h>

// ---------------------------------------------------------------------------
// Transformer encoder layer (post-LN) for gfx1250 (MI455X)
// f16 WMMA + f32 accumulation; TDM (tensor_load_to_lds) tile staging.
// B=2, S=2048, H=1024, N=16 heads, DK=64, DF=4096
// ---------------------------------------------------------------------------

typedef __attribute__((ext_vector_type(16))) _Float16 v16h;
typedef __attribute__((ext_vector_type(8)))  _Float16 v8h;
typedef __attribute__((ext_vector_type(4)))  _Float16 v4h;
typedef __attribute__((ext_vector_type(8)))  float    v8f;
typedef __attribute__((ext_vector_type(4)))  float    v4f;

static constexpr int HDIM  = 1024;
static constexpr int SQ    = 2048;
static constexpr int BATCH = 2;
static constexpr int NHEAD = 16;
static constexpr int DK    = 64;
static constexpr int DF    = 4096;
static constexpr int MROWS = BATCH * SQ;   // 4096
static constexpr int SSTR  = SQ + 8;       // padded score row stride (words)

#if defined(__has_builtin)
#if __has_builtin(__builtin_amdgcn_tensor_load_to_lds) && \
    __has_builtin(__builtin_amdgcn_s_wait_tensorcnt)
#define HAVE_TDM 1
#endif
#endif
#ifndef HAVE_TDM
#define HAVE_TDM 0
#endif

__device__ __forceinline__ v16h pack16(v8h lo, v8h hi) {
  v16h r;
#pragma unroll
  for (int i = 0; i < 8; ++i) { r[i] = lo[i]; r[i + 8] = hi[i]; }
  return r;
}
__device__ __forceinline__ v16h load16(const _Float16* p) {
  return pack16(*(const v8h*)p, *(const v8h*)(p + 8));
}
__device__ __forceinline__ v16h load16_split(const _Float16* row, int ko) {
  return pack16(*(const v8h*)(row + ko), *(const v8h*)(row + 16 + ko));
}
__device__ __forceinline__ v8f wmma16(v16h a, v16h b, v8f c) {
  return __builtin_amdgcn_wmma_f32_16x16x32_f16(false, a, false, b,
                                                (short)0, c, false, false);
}

#if HAVE_TDM
typedef __attribute__((ext_vector_type(4))) unsigned int u32x4;
typedef __attribute__((ext_vector_type(8))) int          i32x8;
typedef __attribute__((ext_vector_type(4))) int          i32x4;

// TDM 2-D tile load: rows x 32 halves, global row stride = krowStride halves,
// LDS padded with 4 DWORDs after every 16 DWORDs (row) -> 40-half LDS stride.
__device__ __forceinline__ void tdm_load_tile(const _Float16* gsrc,
                                              _Float16* lds_dst, int rows,
                                              int kRem, int krowStride) {
  const unsigned long long ga = (unsigned long long)(size_t)gsrc;
  const unsigned lds = (unsigned)(size_t)lds_dst;  // low 32 bits = LDS offset
  u32x4 g0;
  g0[0] = 1u;                                   // count=1, user mode
  g0[1] = lds;                                  // lds_addr
  g0[2] = (unsigned)(ga & 0xFFFFFFFFull);       // global_addr[31:0]
  g0[3] = (unsigned)((ga >> 32) & 0x1FFFFFFull) // global_addr[56:32]
          | (2u << 30);                         // type = 2 (image)
  i32x8 g1;
  // data_size=1 (2B), pad_enable, pad_interval=3 (16 DW), pad_amount=3 (4 DW)
  g1[0] = (int)((1u << 16) | (1u << 20) | (3u << 22) | (3u << 25));
  g1[1] = (int)(((unsigned)kRem & 0xFFFFu) << 16);            // dim0 lo16
  g1[2] = (int)((((unsigned)kRem >> 16) & 0xFFFFu) |
                (((unsigned)rows & 0xFFFFu) << 16));          // dim0 hi, dim1 lo
  g1[3] = (int)((((unsigned)rows >> 16) & 0xFFFFu) |
                (32u << 16));                                 // dim1 hi, tile_dim0=32
  g1[4] = (int)((unsigned)rows & 0xFFFFu);                    // tile_dim1, tile_dim2=0
  g1[5] = krowStride;                                         // dim0_stride[31:0]
  g1[6] = 0;
  g1[7] = 0;
  const i32x4 z4 = {0, 0, 0, 0};
  const i32x8 z8 = {0, 0, 0, 0, 0, 0, 0, 0};
  __builtin_amdgcn_tensor_load_to_lds(g0, g1, z4, z4, z8, 0);
}
#endif

// ---------------------------------------------------------------------------
// GEMM: out[M,N] = act(A[M,K] @ W[N,K]^T + bias); A, W are f16.
// SMODE: 0 -> f32 natural, 1 -> f16 natural, 2 -> f16 transposed-V layout
// ---------------------------------------------------------------------------
template <bool RELU, int SMODE>
__global__ __launch_bounds__(256) void gemm_kernel(
    const _Float16* __restrict__ Ah, const _Float16* __restrict__ Wh,
    const float* __restrict__ bias, void* __restrict__ outp, int M, int N,
    int K) {
  constexpr int BM = 128, BN = 64, BK = 32, LS = 40;
  __shared__ _Float16 As[2][BM * LS];
  __shared__ _Float16 Bs[2][BN * LS];

  const int tid  = threadIdx.x;
  const int lane = tid & 31;
  const int w    = tid >> 5;
  const int wm   = w & 3;   // 4 row strips of 32
  const int wn   = w >> 2;  // 2 col strips of 32
  const int m0   = blockIdx.y * BM;
  const int n0   = blockIdx.x * BN;

  const _Float16* Abase = Ah + (size_t)m0 * K;
  const _Float16* Wbase = Wh + (size_t)n0 * K;

  v8f acc[2][2] = {};
  const int mrow = lane & 15;
  const int ko   = (lane >> 4) * 8;
  const int kb   = (lane >> 4) * 16;

#if HAVE_TDM
  // Wave 0 drives the Tensor Data Mover; double-buffered pipeline.
  if (tid < 32) {
    tdm_load_tile(Abase, As[0], BM, K, K);
    tdm_load_tile(Wbase, Bs[0], BN, K, K);
  }
  int buf = 0;
  for (int kk = 0; kk < K; kk += BK) {
    if (tid < 32) {
      if (kk + BK < K) {
        tdm_load_tile(Abase + kk + BK, As[buf ^ 1], BM, K - (kk + BK), K);
        tdm_load_tile(Wbase + kk + BK, Bs[buf ^ 1], BN, K - (kk + BK), K);
        __builtin_amdgcn_s_wait_tensorcnt(2);  // current tile's 2 loads done
      } else {
        __builtin_amdgcn_s_wait_tensorcnt(0);
      }
    }
    __syncthreads();
    v16h a[2], b[2];
#pragma unroll
    for (int i = 0; i < 2; ++i)
      a[i] = load16_split(&As[buf][(wm * 32 + i * 16 + mrow) * LS], ko);
#pragma unroll
    for (int j = 0; j < 2; ++j)
      b[j] = load16_split(&Bs[buf][(wn * 32 + j * 16 + mrow) * LS], ko);
#pragma unroll
    for (int i = 0; i < 2; ++i)
#pragma unroll
      for (int j = 0; j < 2; ++j) acc[i][j] = wmma16(a[i], b[j], acc[i][j]);
    __syncthreads();
    buf ^= 1;
  }
#else
  // Fallback: cooperative vector staging (all-f16 copies).
  const int arow = tid >> 1, acol = (tid & 1) * 16;
  const int brow = tid >> 2, bcol = (tid & 3) * 8;
  for (int kk = 0; kk < K; kk += BK) {
    {
      const _Float16* src = Abase + (size_t)arow * K + kk + acol;
      __builtin_prefetch(src + BK, 0, 0);
      _Float16* dst = &As[0][arow * LS + acol];
      *(v8h*)dst = *(const v8h*)src;
      *(v8h*)(dst + 8) = *(const v8h*)(src + 8);
    }
    {
      const _Float16* src = Wbase + (size_t)brow * K + kk + bcol;
      __builtin_prefetch(src + BK, 0, 0);
      *(v8h*)&Bs[0][brow * LS + bcol] = *(const v8h*)src;
    }
    __syncthreads();
    v16h a[2], b[2];
#pragma unroll
    for (int i = 0; i < 2; ++i)
      a[i] = load16_split(&As[0][(wm * 32 + i * 16 + mrow) * LS], ko);
#pragma unroll
    for (int j = 0; j < 2; ++j)
      b[j] = load16_split(&Bs[0][(wn * 32 + j * 16 + mrow) * LS], ko);
#pragma unroll
    for (int i = 0; i < 2; ++i)
#pragma unroll
      for (int j = 0; j < 2; ++j) acc[i][j] = wmma16(a[i], b[j], acc[i][j]);
    __syncthreads();
  }
#endif

  // ---- epilogue: bias / relu / store ----
  const int mwave = m0 + wm * 32;
  const int nwave = n0 + wn * 32;
#pragma unroll
  for (int j = 0; j < 2; ++j) {
    const int n = nwave + j * 16 + (lane & 15);
    const float bv = bias ? bias[n] : 0.0f;
#pragma unroll
    for (int i = 0; i < 2; ++i) {
      const int mbase = mwave + i * 16 + (lane >> 4) * 8;
#pragma unroll
      for (int r = 0; r < 8; ++r) {
        float v = acc[i][j][r] + bv;
        if (RELU) v = fmaxf(v, 0.0f);
        const int m = mbase + r;
        if (SMODE == 0) {
          ((float*)outp)[(size_t)m * N + n] = v;
        } else if (SMODE == 1) {
          ((_Float16*)outp)[(size_t)m * N + n] = (_Float16)v;
        } else {  // transposed V: [(b*16+head)*64+dk][t]
          const int bb = m >> 11, t = m & (SQ - 1);
          const int head = n >> 6, dk = n & 63;
          ((_Float16*)outp)[(((size_t)bb * NHEAD + head) * DK + dk) * SQ + t] =
              (_Float16)v;
        }
      }
    }
  }
}

// ---------------------------------------------------------------------------
// Attention: one block per (b, head, 32-query rows). 32x2048 f32 scores in
// ~257KB dynamic LDS (CDNA5 WGP has 320KB).
// ---------------------------------------------------------------------------
__global__ __launch_bounds__(256) void attn_kernel(
    const _Float16* __restrict__ qh, const _Float16* __restrict__ kh,
    const _Float16* __restrict__ vt, _Float16* __restrict__ outh, float scale) {
  extern __shared__ float sc[];  // [32][SSTR]
  const int tid  = threadIdx.x;
  const int lane = tid & 31;
  const int w    = tid >> 5;
  const int tb   = blockIdx.x & 63;
  const int bh   = blockIdx.x >> 6;  // b*16 + head
  const int b    = bh >> 4;
  const int h    = bh & 15;
  const int t0   = tb * 32;

  const int mrow = lane & 15;
  const int ko   = (lane >> 4) * 8;
  const int kb16 = (lane >> 4) * 16;

  // ---- phase 1: scores = (q @ k^T) * scale ----
  v16h aq[2][2];
#pragma unroll
  for (int mi = 0; mi < 2; ++mi) {
    const _Float16* qrow =
        qh + (size_t)(b * SQ + t0 + mi * 16 + mrow) * HDIM + h * DK;
#pragma unroll
    for (int ks = 0; ks < 2; ++ks) aq[mi][ks] = load16_split(qrow + ks * 32, ko);
  }
  for (int s0 = w * 32; s0 < SQ; s0 += 256) {
    v8f acc[2][2] = {};
#pragma unroll
    for (int ks = 0; ks < 2; ++ks) {
      v16h bf[2];
#pragma unroll
      for (int sj = 0; sj < 2; ++sj) {
        const _Float16* p = kh +
            (size_t)(b * SQ + s0 + sj * 16 + mrow) * HDIM + h * DK + ks * 32;
        bf[sj] = load16_split(p, ko);
      }
#pragma unroll
      for (int mi = 0; mi < 2; ++mi)
#pragma unroll
        for (int sj = 0; sj < 2; ++sj)
          acc[mi][sj] = wmma16(aq[mi][ks], bf[sj], acc[mi][sj]);
    }
#pragma unroll
    for (int mi = 0; mi < 2; ++mi)
#pragma unroll
      for (int sj = 0; sj < 2; ++sj) {
        const int mB = mi * 16 + (lane >> 4) * 8;
        const int s  = s0 + sj * 16 + (lane & 15);
#pragma unroll
        for (int r = 0; r < 8; ++r)
          sc[(mB + r) * SSTR + s] = acc[mi][sj][r] * scale;
      }
  }
  __syncthreads();

  // ---- phase 2: softmax over s (8 lanes per row, wave32 shuffles) ----
  {
    const int row = tid >> 3;
    const int j   = tid & 7;
    float* prow = sc + row * SSTR;
    float mx = -3.0e38f;
    for (int s = j; s < SQ; s += 8) mx = fmaxf(mx, prow[s]);
    mx = fmaxf(mx, __shfl_xor(mx, 1));
    mx = fmaxf(mx, __shfl_xor(mx, 2));
    mx = fmaxf(mx, __shfl_xor(mx, 4));
    float sum = 0.0f;
    for (int s = j; s < SQ; s += 8) {
      const float e = __expf(prow[s] - mx);
      prow[s] = e;
      sum += e;
    }
    sum += __shfl_xor(sum, 1);
    sum += __shfl_xor(sum, 2);
    sum += __shfl_xor(sum, 4);
    const float inv = 1.0f / sum;
    for (int s = j; s < SQ; s += 8) prow[s] *= inv;
  }
  __syncthreads();

  // ---- phase 3: out = p @ v ; each wave one 16x16 tile ----
  {
    const int mi = w & 1;
    const int nj = w >> 1;
    v8f acc = {};
    const _Float16* vbase = vt + ((size_t)bh * DK + nj * 16 + (lane & 15)) * SQ;
    const float* arow = sc + (mi * 16 + mrow) * SSTR;
    for (int kk = 0; kk < SQ; kk += 32) {
      v16h a;
      const float* p = arow + kk;
#pragma unroll
      for (int e = 0; e < 8; ++e) {
        a[e]     = (_Float16)p[ko + e];
        a[e + 8] = (_Float16)p[16 + ko + e];
      }
      const v16h bf = load16_split(vbase + kk, ko);
      acc = wmma16(a, bf, acc);
    }
    const int ncol  = h * DK + nj * 16 + (lane & 15);
    const int mbase = t0 + mi * 16 + (lane >> 4) * 8;
#pragma unroll
    for (int r = 0; r < 8; ++r)
      outh[(size_t)(b * SQ + mbase + r) * HDIM + ncol] = (_Float16)acc[r];
  }
}

// ---------------------------------------------------------------------------
// out = LayerNorm(x + y) * g + b  (+ optional f16 mirror for next GEMM)
// ---------------------------------------------------------------------------
__global__ __launch_bounds__(256) void add_ln_kernel(
    const float* __restrict__ x, const float* __restrict__ y,
    const float* __restrict__ g, const float* __restrict__ be,
    float* __restrict__ out, _Float16* __restrict__ outh) {
  __shared__ float red[16];
  const int row = blockIdx.x;
  const int tid = threadIdx.x;
  const float* xr = x + (size_t)row * HDIM;
  const float* yr = y + (size_t)row * HDIM;
  const v4f xv = *(const v4f*)(xr + tid * 4);
  const v4f yv = *(const v4f*)(yr + tid * 4);
  float v[4];
  float s = 0.0f, s2 = 0.0f;
#pragma unroll
  for (int i = 0; i < 4; ++i) {
    v[i] = xv[i] + yv[i];
    s += v[i];
    s2 += v[i] * v[i];
  }
#pragma unroll
  for (int o = 1; o < 32; o <<= 1) {
    s  += __shfl_xor(s, o);
    s2 += __shfl_xor(s2, o);
  }
  if ((tid & 31) == 0) { red[tid >> 5] = s; red[8 + (tid >> 5)] = s2; }
  __syncthreads();
  float ts = 0.0f, ts2 = 0.0f;
#pragma unroll
  for (int i = 0; i < 8; ++i) { ts += red[i]; ts2 += red[8 + i]; }
  const float mu  = ts * (1.0f / HDIM);
  const float var = ts2 * (1.0f / HDIM) - mu * mu;
  const float rs  = rsqrtf(var + 1e-5f);
  float o4[4];
#pragma unroll
  for (int i = 0; i < 4; ++i) {
    const int c = tid * 4 + i;
    o4[i] = (v[i] - mu) * rs * g[c] + be[c];
  }
  *(v4f*)(out + (size_t)row * HDIM + tid * 4) = {o4[0], o4[1], o4[2], o4[3]};
  if (outh) {
    v4h h = {(_Float16)o4[0], (_Float16)o4[1], (_Float16)o4[2], (_Float16)o4[3]};
    *(v4h*)(outh + (size_t)row * HDIM + tid * 4) = h;
  }
}

// ---------------------------------------------------------------------------
// f32 -> f16 bulk convert (n must be a multiple of 4)
// ---------------------------------------------------------------------------
__global__ __launch_bounds__(256) void cvt_f16_kernel(
    const float* __restrict__ in, _Float16* __restrict__ out, int n4) {
  const int i = blockIdx.x * 256 + threadIdx.x;
  if (i < n4) {
    const v4f f = *(const v4f*)(in + (size_t)i * 4);
    v4h h = {(_Float16)f[0], (_Float16)f[1], (_Float16)f[2], (_Float16)f[3]};
    *(v4h*)(out + (size_t)i * 4) = h;
  }
}

// ---------------------------------------------------------------------------
extern "C" void kernel_launch(void* const* d_in, const int* in_sizes, int n_in,
                              void* d_out, int out_size, void* d_ws,
                              size_t ws_size, hipStream_t stream) {
  const float* x     = (const float*)d_in[0];
  const float* Wq    = (const float*)d_in[1];
  const float* bq    = (const float*)d_in[2];
  const float* Wk    = (const float*)d_in[3];
  const float* bk    = (const float*)d_in[4];
  const float* Wv    = (const float*)d_in[5];
  const float* bv    = (const float*)d_in[6];
  const float* Wo    = (const float*)d_in[7];
  const float* bo    = (const float*)d_in[8];
  const float* W1    = (const float*)d_in[9];
  const float* b1    = (const float*)d_in[10];
  const float* W2    = (const float*)d_in[11];
  const float* b2    = (const float*)d_in[12];
  const float* ln1_g = (const float*)d_in[13];
  const float* ln1_b = (const float*)d_in[14];
  const float* ln2_g = (const float*)d_in[15];
  const float* ln2_b = (const float*)d_in[16];
  float* out = (float*)d_out;

  char* ws = (char*)d_ws;
  size_t off = 0;
  auto take = [&](size_t bytes) {
    void* p = ws + off;
    off += (bytes + 255) & ~size_t(255);
    return p;
  };
  _Float16* q_h    = (_Float16*)take((size_t)MROWS * HDIM * 2);
  _Float16* k_h    = (_Float16*)take((size_t)MROWS * HDIM * 2);
  _Float16* v_t    = (_Float16*)take((size_t)MROWS * HDIM * 2);
  _Float16* attn_h = (_Float16*)take((size_t)MROWS * HDIM * 2);
  float*    proj   = (float*)take((size_t)MROWS * HDIM * 4);
  float*    hbuf   = (float*)take((size_t)MROWS * HDIM * 4);
  _Float16* hbuf_h = (_Float16*)take((size_t)MROWS * HDIM * 2);
  _Float16* ff1_h  = (_Float16*)take((size_t)MROWS * DF * 2);
  float*    ff2    = (float*)take((size_t)MROWS * HDIM * 4);
  _Float16* x_h    = (_Float16*)take((size_t)MROWS * HDIM * 2);
  _Float16* wq_h   = (_Float16*)take((size_t)HDIM * HDIM * 2);
  _Float16* wk_h   = (_Float16*)take((size_t)HDIM * HDIM * 2);
  _Float16* wv_h   = (_Float16*)take((size_t)HDIM * HDIM * 2);
  _Float16* wo_h   = (_Float16*)take((size_t)HDIM * HDIM * 2);
  _Float16* w1_h   = (_Float16*)take((size_t)DF * HDIM * 2);
  _Float16* w2_h   = (_Float16*)take((size_t)HDIM * DF * 2);

  const dim3 blk(256);
  auto cvt = [&](const float* src, _Float16* dst, size_t n) {
    const int n4 = (int)(n / 4);
    cvt_f16_kernel<<<dim3((n4 + 255) / 256), blk, 0, stream>>>(src, dst, n4);
  };
  cvt(x, x_h, (size_t)MROWS * HDIM);
  cvt(Wq, wq_h, (size_t)HDIM * HDIM);
  cvt(Wk, wk_h, (size_t)HDIM * HDIM);
  cvt(Wv, wv_h, (size_t)HDIM * HDIM);
  cvt(Wo, wo_h, (size_t)HDIM * HDIM);
  cvt(W1, w1_h, (size_t)DF * HDIM);
  cvt(W2, w2_h, (size_t)HDIM * DF);

  const dim3 gH(HDIM / 64, MROWS / 128);
  const dim3 gF(DF / 64, MROWS / 128);

  // QKV projections (f16 WMMA, f16 outputs; V transposed)
  gemm_kernel<false, 1><<<gH, blk, 0, stream>>>(x_h, wq_h, bq, q_h, MROWS, HDIM, HDIM);
  gemm_kernel<false, 1><<<gH, blk, 0, stream>>>(x_h, wk_h, bk, k_h, MROWS, HDIM, HDIM);
  gemm_kernel<false, 2><<<gH, blk, 0, stream>>>(x_h, wv_h, bv, v_t, MROWS, HDIM, HDIM);

  // Attention (scores kept in 257 KB dynamic LDS per workgroup)
  const size_t smem = (size_t)32 * SSTR * sizeof(float);
  (void)hipFuncSetAttribute((const void*)attn_kernel,
                            hipFuncAttributeMaxDynamicSharedMemorySize,
                            (int)smem);
  attn_kernel<<<dim3(BATCH * NHEAD * (SQ / 32)), blk, smem, stream>>>(
      q_h, k_h, v_t, attn_h, 0.125f);

  // Output projection + residual LN1 (also emits f16 mirror for FFN1)
  gemm_kernel<false, 0><<<gH, blk, 0, stream>>>(attn_h, wo_h, bo, proj, MROWS, HDIM, HDIM);
  add_ln_kernel<<<dim3(MROWS), blk, 0, stream>>>(x, proj, ln1_g, ln1_b, hbuf, hbuf_h);

  // FFN
  gemm_kernel<true, 1><<<gF, blk, 0, stream>>>(hbuf_h, w1_h, b1, ff1_h, MROWS, DF, HDIM);
  gemm_kernel<false, 0><<<gH, blk, 0, stream>>>(ff1_h, w2_h, b2, ff2, MROWS, HDIM, DF);
  add_ln_kernel<<<dim3(MROWS), blk, 0, stream>>>(hbuf, ff2, ln2_g, ln2_b, out, nullptr);
}
